// AttentionSubsample_31078383354064
// MI455X (gfx1250) — matrix operand order repaired and checked by reading, not compiled, not run on hardware
//
#include <hip/hip_runtime.h>
#include <hip/hip_bf16.h>
#include <stdint.h>

typedef __attribute__((ext_vector_type(16))) _Float16 v16h;
typedef __attribute__((ext_vector_type(8)))  _Float16 v8h;
typedef __attribute__((ext_vector_type(8)))  float    v8f;

constexpr int NB  = 2;
constexpr int T   = 16;
constexpr int S   = 196;
constexpr int D   = 768;
constexpr int H   = 12;
constexpr int HD  = 64;
constexpr int QL  = T * S;        // 3136 query rows per batch
constexpr int KVL = T * (S / 2);  // 1568 key/value rows per batch
constexpr int CQKV = 3 * D;       // 2304
constexpr float ATT_SCALE = 0.125f; // (768/12)^-0.5

// ---------------------------------------------------------------- helpers
__device__ __forceinline__ v8f zero8f() {
  v8f z;
#pragma unroll
  for (int i = 0; i < 8; ++i) z[i] = 0.0f;
  return z;
}

// D = A(16x32 f16) * B(32x16 f16) + C(16x16 f32)
__device__ __forceinline__ v8f wmma16(v16h a, v16h b, v8f c) {
  return __builtin_amdgcn_wmma_f32_16x16x32_f16(false, a, false, b,
                                                (short)0, c, false, false);
}

// A fragment, 16x32 f16, ISA layout: lane<16 -> K = kb..kb+7, kb+16..kb+23
// with kb=0; lane>=16 -> kb=8. Row M = lane&15. Two b128 loads per lane.
__device__ __forceinline__ v16h load_a_frag(const _Float16* base, int ld,
                                            int mbase, int kbase) {
  const int lane = threadIdx.x & 31;
  const _Float16* p = base + (size_t)(mbase + (lane & 15)) * ld + kbase +
                      ((lane & 16) ? 8 : 0);
  v8h lo = *(const v8h*)p;
  v8h hi = *(const v8h*)(p + 16);
  v16h r;
#pragma unroll
  for (int i = 0; i < 8; ++i) { r[i] = lo[i]; r[i + 8] = hi[i]; }
  return r;
}

// B fragment from transposed storage Bt[N][K]: lane N = lane&15,
// K = kb..kb+7, kb+16..kb+23, kb = lane<16 ? 0 : 8. Two b128 loads.
__device__ __forceinline__ v16h load_b_frag(const _Float16* base, int ld,
                                            int nbase, int kbase) {
  const int lane = threadIdx.x & 31;
  const _Float16* p = base + (size_t)(nbase + (lane & 15)) * ld + kbase +
                      ((lane & 16) ? 8 : 0);
  v8h lo = *(const v8h*)p;
  v8h hi = *(const v8h*)(p + 16);
  v16h r;
#pragma unroll
  for (int i = 0; i < 8; ++i) { r[i] = lo[i]; r[i + 8] = hi[i]; }
  return r;
}

// CDNA5 async copy global->LDS (ASYNCcnt path), 16 bytes per lane.
__device__ __forceinline__ void async_copy_b128(const _Float16* lds_dst,
                                                const _Float16* gsrc) {
  unsigned lds_off = (unsigned)(uintptr_t)lds_dst;          // low 32b = LDS addr
  unsigned long long ga = (unsigned long long)(uintptr_t)gsrc;
  asm volatile("global_load_async_to_lds_b128 %0, %1, off"
               :: "v"(lds_off), "v"(ga) : "memory");
}
__device__ __forceinline__ void wait_async_le4() {
  asm volatile("s_wait_asynccnt 4" ::: "memory");
}
__device__ __forceinline__ void wait_async_le0() {
  asm volatile("s_wait_asynccnt 0" ::: "memory");
}

// ---------------------------------------------------------------- K1: QKV
// C[6272x2304] = X[6272x768](f32->f16) @ Wqkv[768x2304](f32->f16),
// scatter: Q (all rows), K (rows s even, compacted), V (transposed per head).
__global__ __launch_bounds__(256) void qkv_kernel(
    const float* __restrict__ X, const float* __restrict__ W,
    _Float16* __restrict__ Qh, _Float16* __restrict__ Kh,
    _Float16* __restrict__ Vt) {
  constexpr int LDA = 40, LDB = 40;
  __shared__ _Float16 As[128 * LDA];   // A tile 128x32 (row-major)
  __shared__ _Float16 Bs[128 * LDB];   // B tile transposed: [col][k]
  const int n  = blockIdx.z;
  const int r0 = blockIdx.y * 128;
  const int c0 = blockIdx.x * 128;
  const int tid  = threadIdx.x;
  const int wave = tid >> 5;
  const int wm = (wave & 3) * 32;
  const int wn = (wave >> 2) * 64;
  const int lane = tid & 31;
  const int ln15 = lane & 15;
  const int hi8  = (lane & 16) ? 8 : 0;

  v8f acc[2][4];
#pragma unroll
  for (int i = 0; i < 2; ++i)
#pragma unroll
    for (int j = 0; j < 4; ++j) acc[i][j] = zero8f();

  const float* Xn = X + (size_t)n * QL * D;

  for (int k0 = 0; k0 < D; k0 += 32) {
    // prefetch next k-slice (global_prefetch_b8)
    if (k0 + 32 < D) {
      int pr = r0 + (tid >> 3);
      pr = pr < QL ? pr : QL - 1;
      __builtin_prefetch(Xn + (size_t)pr * D + k0 + 32 + (tid & 7) * 4, 0, 1);
      __builtin_prefetch(W + (size_t)(k0 + 32 + (tid >> 5)) * CQKV + c0 +
                             (tid & 31) * 4, 0, 1);
    }
    // Batch all global loads first (row-clamped: duplicate rows only feed
    // output rows that are discarded at store time), then convert.
    float4 av[4], bw[4];
#pragma unroll
    for (int i = 0; i < 4; ++i) {
      int slot = tid + i * 256;
      int r = slot >> 3, c4 = (slot & 7) * 4;
      int grow = r0 + r;
      int crow = grow < QL ? grow : QL - 1;
      av[i] = *(const float4*)(Xn + (size_t)crow * D + k0 + c4);
    }
#pragma unroll
    for (int i = 0; i < 4; ++i) {
      int slot = tid + i * 256;
      int kk = slot >> 5, c4 = (slot & 31) * 4;
      bw[i] = *(const float4*)(W + (size_t)(k0 + kk) * CQKV + c0 + c4);
    }
#pragma unroll
    for (int i = 0; i < 4; ++i) {
      int slot = tid + i * 256;
      int r = slot >> 3, c4 = (slot & 7) * 4;
      _Float16* dst = &As[r * LDA + c4];
      dst[0] = (_Float16)av[i].x; dst[1] = (_Float16)av[i].y;
      dst[2] = (_Float16)av[i].z; dst[3] = (_Float16)av[i].w;
    }
#pragma unroll
    for (int i = 0; i < 4; ++i) {
      int slot = tid + i * 256;
      int kk = slot >> 5, c4 = (slot & 31) * 4;
      Bs[(c4 + 0) * LDB + kk] = (_Float16)bw[i].x;
      Bs[(c4 + 1) * LDB + kk] = (_Float16)bw[i].y;
      Bs[(c4 + 2) * LDB + kk] = (_Float16)bw[i].z;
      Bs[(c4 + 3) * LDB + kk] = (_Float16)bw[i].w;
    }
    __syncthreads();
    v16h a0 = load_a_frag(As, LDA, wm, 0);
    v16h a1 = load_a_frag(As, LDA, wm + 16, 0);
#pragma unroll
    for (int nt = 0; nt < 4; ++nt) {
      v16h b = load_b_frag(Bs, LDB, wn + nt * 16, 0);
      acc[0][nt] = wmma16(a0, b, acc[0][nt]);
      acc[1][nt] = wmma16(a1, b, acc[1][nt]);
    }
    __syncthreads();
  }

  // Scatter f16 results into Q / subsampled K / transposed subsampled V.
#pragma unroll
  for (int mt = 0; mt < 2; ++mt)
#pragma unroll
    for (int nt = 0; nt < 4; ++nt)
#pragma unroll
      for (int r = 0; r < 8; ++r) {
        int row = r0 + wm + mt * 16 + r + hi8;
        if (row >= QL) continue;
        int col = c0 + wn + nt * 16 + ln15;
        _Float16 hv = (_Float16)acc[mt][nt][r];
        int t = row / S, s = row - t * S;
        if (col < D) {
          Qh[((size_t)n * QL + row) * D + col] = hv;
        } else if (col < 2 * D) {
          if ((s & 1) == 0)
            Kh[((size_t)n * KVL + t * (S / 2) + (s >> 1)) * D + (col - D)] = hv;
        } else {
          if ((s & 1) == 0) {
            int dc = col - 2 * D;
            Vt[(((size_t)n * H + (dc >> 6)) * HD + (dc & 63)) * KVL +
               t * (S / 2) + (s >> 1)] = hv;
          }
        }
      }
}

// ---------------------------------------------------------------- K2: flash attention
// grid(49, 12, 2), block 128 = 4 waves; wave owns 16 query rows.
// 49 KV tiles of 32 keys; double-buffered async LDS copies overlap with WMMA.
__global__ __launch_bounds__(128) void attn_kernel(
    const _Float16* __restrict__ Qh, const _Float16* __restrict__ Kh,
    const _Float16* __restrict__ Vt, _Float16* __restrict__ Og) {
  constexpr int LDK = 72, LDV = 40, LDP = 40;
  __shared__ _Float16 Ks[2][32 * LDK];    // K tile row-major = Bt for Q*K^T
  __shared__ _Float16 Vs[2][64 * LDV];    // V^T tile [d][key] = Bt for P*V
  __shared__ _Float16 Ps[4 * 16 * LDP];   // per-wave P staging (C->A relayout)
  const int n = blockIdx.z, h = blockIdx.y;
  const int tid = threadIdx.x;
  const int wave = tid >> 5;
  const int lane = tid & 31;
  const int ln15 = lane & 15;
  const int hi8  = (lane & 16) ? 8 : 0;
  const int q0 = blockIdx.x * 64 + wave * 16;
  _Float16* Psw = &Ps[wave * 16 * LDP];

  const _Float16* Qbase = Qh + ((size_t)n * QL + q0) * D + h * HD;
  v16h qf0 = load_a_frag(Qbase, D, 0, 0);    // d = 0..31
  v16h qf1 = load_a_frag(Qbase, D, 0, 32);   // d = 32..63

  v8f o[4];
#pragma unroll
  for (int i = 0; i < 4; ++i) o[i] = zero8f();
  float m8[8], l8[8];
#pragma unroll
  for (int r = 0; r < 8; ++r) { m8[r] = -1e30f; l8[r] = 0.0f; }

  const _Float16* Kbase = Kh + (size_t)n * KVL * D + h * HD;
  const _Float16* Vbase = Vt + ((size_t)n * H + h) * (size_t)HD * KVL;

  // 4 async b128 copies per thread per tile (2 K chunks + 2 V chunks).
  auto issue_tile = [&](int kv0, int buf) {
#pragma unroll
    for (int i = 0; i < 2; ++i) {
      int chunk = tid + i * 128;
      int row = chunk >> 3, c8 = (chunk & 7) * 8;
      async_copy_b128(&Ks[buf][row * LDK + c8],
                      Kbase + (size_t)(kv0 + row) * D + c8);
    }
#pragma unroll
    for (int i = 0; i < 2; ++i) {
      int chunk = tid + i * 128;
      int row = chunk >> 2, c8 = (chunk & 3) * 8;
      async_copy_b128(&Vs[buf][row * LDV + c8],
                      Vbase + (size_t)row * KVL + kv0 + c8);
    }
  };

  constexpr int NT = KVL / 32;   // 49
  issue_tile(0, 0);

  for (int it = 0; it < NT; ++it) {
    const int cur = it & 1;
    if (it + 1 < NT) {
      // Safe: end-of-iteration barrier below guarantees every wave finished
      // reading buf cur^1 (tile it-1) before these copies overwrite it.
      issue_tile((it + 1) * 32, cur ^ 1);
      wait_async_le4();  // in-order completion => tile `it`'s 4 copies done
    } else {
      wait_async_le0();
    }
    __syncthreads();
    const _Float16* Kst = Ks[cur];
    const _Float16* Vst = Vs[cur];

    // S(16x32) = Q(16x64) * K^T : 2 key tiles x 2 d-chunks
    v8f s0 = zero8f(), s1 = zero8f();
    {
      v16h b00 = load_b_frag(Kst, LDK, 0,  0);
      v16h b10 = load_b_frag(Kst, LDK, 16, 0);
      s0 = wmma16(qf0, b00, s0);
      s1 = wmma16(qf0, b10, s1);
      v16h b01 = load_b_frag(Kst, LDK, 0,  32);
      v16h b11 = load_b_frag(Kst, LDK, 16, 32);
      s0 = wmma16(qf1, b01, s0);
      s1 = wmma16(qf1, b11, s1);
    }

    // Online softmax: rows live across the 16-lane half -> xor-shuffle reduce.
#pragma unroll
    for (int r = 0; r < 8; ++r) {
      float a = s0[r] * ATT_SCALE;
      float b = s1[r] * ATT_SCALE;
      float mx = fmaxf(a, b);
#pragma unroll
      for (int off = 1; off < 16; off <<= 1)
        mx = fmaxf(mx, __shfl_xor(mx, off, 32));
      float mnew = fmaxf(m8[r], mx);
      float corr = __expf(m8[r] - mnew);
      float pa = __expf(a - mnew);
      float pb = __expf(b - mnew);
      float rs = pa + pb;
#pragma unroll
      for (int off = 1; off < 16; off <<= 1)
        rs += __shfl_xor(rs, off, 32);
      l8[r] = l8[r] * corr + rs;
      m8[r] = mnew;
#pragma unroll
      for (int dt = 0; dt < 4; ++dt) o[dt][r] *= corr;
      Psw[(r + hi8) * LDP + ln15]      = (_Float16)pa;   // keys 0..15
      Psw[(r + hi8) * LDP + 16 + ln15] = (_Float16)pb;   // keys 16..31
    }
    __syncthreads();   // order P stores (cross-lane) before A-frag reads

    // O(16x64) += P(16x32) * V(32x64)
    v16h pf = load_a_frag(Psw, LDP, 0, 0);
#pragma unroll
    for (int dt = 0; dt < 4; ++dt) {
      v16h bv = load_b_frag(Vst, LDV, dt * 16, 0);
      o[dt] = wmma16(pf, bv, o[dt]);
    }
    __syncthreads();   // all reads of buf[cur] done before it is refilled
  }

#pragma unroll
  for (int r = 0; r < 8; ++r) {
    float inv = 1.0f / l8[r];
    int row = q0 + r + hi8;
#pragma unroll
    for (int dt = 0; dt < 4; ++dt)
      Og[((size_t)n * QL + row) * D + h * HD + dt * 16 + ln15] =
          (_Float16)(o[dt][r] * inv);
  }
}

// ---------------------------------------------------------------- K3: out proj
// out[6272x768] = Og(f16) @ Wproj(f32->f16) + b_proj, f32 output.
__global__ __launch_bounds__(256) void proj_kernel(
    const _Float16* __restrict__ A, const float* __restrict__ W,
    const float* __restrict__ bias, float* __restrict__ out) {
  constexpr int LDA = 40, LDB = 40;
  __shared__ _Float16 As[128 * LDA];
  __shared__ _Float16 Bs[128 * LDB];
  const int n  = blockIdx.z;
  const int r0 = blockIdx.y * 128;
  const int c0 = blockIdx.x * 128;
  const int tid  = threadIdx.x;
  const int wave = tid >> 5;
  const int wm = (wave & 3) * 32;
  const int wn = (wave >> 2) * 64;
  const int lane = tid & 31;
  const int ln15 = lane & 15;
  const int hi8  = (lane & 16) ? 8 : 0;

  v8f acc[2][4];
#pragma unroll
  for (int i = 0; i < 2; ++i)
#pragma unroll
    for (int j = 0; j < 4; ++j) acc[i][j] = zero8f();

  const _Float16* An = A + (size_t)n * QL * D;

  for (int k0 = 0; k0 < D; k0 += 32) {
    if (k0 + 32 < D)
      __builtin_prefetch(W + (size_t)(k0 + 32 + (tid >> 5)) * D + c0 +
                             (tid & 31) * 4, 0, 1);
    // A tile (f16 already): async global->LDS, row-clamped (dup rows feed
    // only discarded output rows). 512 b128 chunks, 2 per thread.
#pragma unroll
    for (int i = 0; i < 2; ++i) {
      int slot = tid + i * 256;
      int r = slot >> 2, c8 = (slot & 3) * 8;
      int grow = r0 + r;
      int crow = grow < QL ? grow : QL - 1;
      async_copy_b128(&As[r * LDA + c8],
                      An + (size_t)crow * D + k0 + c8);
    }
    // B tile f32 -> f16, transposed store; overlaps with async A copies.
    float4 bw[4];
#pragma unroll
    for (int i = 0; i < 4; ++i) {
      int slot = tid + i * 256;
      int kk = slot >> 5, c4 = (slot & 31) * 4;
      bw[i] = *(const float4*)(W + (size_t)(k0 + kk) * D + c0 + c4);
    }
#pragma unroll
    for (int i = 0; i < 4; ++i) {
      int slot = tid + i * 256;
      int kk = slot >> 5, c4 = (slot & 31) * 4;
      Bs[(c4 + 0) * LDB + kk] = (_Float16)bw[i].x;
      Bs[(c4 + 1) * LDB + kk] = (_Float16)bw[i].y;
      Bs[(c4 + 2) * LDB + kk] = (_Float16)bw[i].z;
      Bs[(c4 + 3) * LDB + kk] = (_Float16)bw[i].w;
    }
    wait_async_le0();
    __syncthreads();
    v16h a0 = load_a_frag(As, LDA, wm, 0);
    v16h a1 = load_a_frag(As, LDA, wm + 16, 0);
#pragma unroll
    for (int nt = 0; nt < 4; ++nt) {
      v16h b = load_b_frag(Bs, LDB, wn + nt * 16, 0);
      acc[0][nt] = wmma16(a0, b, acc[0][nt]);
      acc[1][nt] = wmma16(a1, b, acc[1][nt]);
    }
    __syncthreads();
  }

#pragma unroll
  for (int mt = 0; mt < 2; ++mt)
#pragma unroll
    for (int nt = 0; nt < 4; ++nt)
#pragma unroll
      for (int r = 0; r < 8; ++r) {
        int row = r0 + wm + mt * 16 + r + hi8;
        if (row >= QL) continue;
        int col = c0 + wn + nt * 16 + ln15;
        out[((size_t)n * QL + row) * D + col] = acc[mt][nt][r] + bias[col];
      }
}

// ---------------------------------------------------------------- launch
extern "C" void kernel_launch(void* const* d_in, const int* in_sizes, int n_in,
                              void* d_out, int out_size, void* d_ws,
                              size_t ws_size, hipStream_t stream) {
  const float* x     = (const float*)d_in[0];
  const float* Wqkv  = (const float*)d_in[1];
  const float* Wproj = (const float*)d_in[2];
  const float* bproj = (const float*)d_in[3];
  float* out = (float*)d_out;

  _Float16* ws = (_Float16*)d_ws;
  _Float16* Qh = ws;                                   // NB*QL*D
  _Float16* Kh = Qh + (size_t)NB * QL * D;             // NB*KVL*D
  _Float16* Vt = Kh + (size_t)NB * KVL * D;            // NB*H*HD*KVL
  _Float16* Og = Vt + (size_t)NB * H * HD * KVL;       // NB*QL*D

  dim3 g1(CQKV / 128, (QL + 127) / 128, NB);           // (18, 25, 2)
  qkv_kernel<<<g1, dim3(256), 0, stream>>>(x, Wqkv, Qh, Kh, Vt);

  dim3 g2(QL / 64, H, NB);                             // (49, 12, 2)
  attn_kernel<<<g2, dim3(128), 0, stream>>>(Qh, Kh, Vt, Og);

  dim3 g3(D / 128, (QL + 127) / 128, NB);              // (6, 25, 2)
  proj_kernel<<<g3, dim3(256), 0, stream>>>(Og, Wproj, bproj, out);
}
